// TemporalSelfAttentionv2_66417374265739
// MI455X (gfx1250) — matrix operand order repaired and checked by reading, not compiled, not run on hardware
//
#include <hip/hip_runtime.h>
#include <math.h>
#include <stdint.h>

typedef __attribute__((ext_vector_type(16))) _Float16 v16h;
typedef __attribute__((ext_vector_type(8)))  float    v8f;

#define B_ 16
#define T_ 64
#define N_ 207
#define F_ 256
#define KSTEPS 8     // 256 / 32
#define NTILES 16    // 256 / 16
#define XROW_DW 260  // 256 + 4 DW pad: row stride = 4 banks mod 64 -> minimal DS conflicts

// Packed B-fragment layout: [kstep][ntile][lane][16 halves]; each lane's 16
// halves contiguous (32B) so a B fragment is one vector load (stays L2-resident).
__device__ __forceinline__ int wpk_off(int s, int j, int lane) {
    return ((s * NTILES + j) * 32 + lane) * 16;
}

__device__ __forceinline__ float gelu_exact(float x) {
    return 0.5f * x * (1.0f + erff(x * 0.70710678118654752f));
}

// Async global -> LDS copy, 16B per lane, tracked by ASYNCcnt (CDNA5 path).
__device__ __forceinline__ void async_copy_b128(uint32_t lds_off, const void* gsrc) {
    asm volatile("global_load_async_to_lds_b128 %0, %1, off"
                 :: "v"(lds_off), "v"((unsigned long long)(uintptr_t)gsrc)
                 : "memory");
}

__device__ __forceinline__ void wait_asynccnt0() {
    asm volatile("s_wait_asynccnt 0x0" ::: "memory");
}

// ---------------------------------------------------------------------------
// Kernel 0: pack W (f32 row-major [K=256][N=256]) into f16 WMMA B-fragments.
// B layout per ISA: lane l holds col N=(l&15), K = (l>>4)*8 + (i&7) + (i>>3)*16.
// ---------------------------------------------------------------------------
__global__ __launch_bounds__(256)
void pack_w_kernel(const float* __restrict__ Wh, const float* __restrict__ Wa,
                   _Float16* __restrict__ WhP, _Float16* __restrict__ WaP) {
    int tid = blockIdx.x * blockDim.x + threadIdx.x;   // 0..16383
    const float* src = (tid < 8192) ? Wh : Wa;
    _Float16*    dst = (tid < 8192) ? WhP : WaP;
    int t    = tid & 8191;
    int lane = t & 31;
    int j    = (t >> 5) & 15;
    int s    = t >> 9;
    int col  = j * 16 + (lane & 15);
    int krow = s * 32 + (lane >> 4) * 8;
    _Float16* d = dst + wpk_off(s, j, lane);
#pragma unroll
    for (int i = 0; i < 16; ++i)
        d[i] = (_Float16)src[(size_t)(krow + (i & 7) + (i >> 3) * 16) * F_ + col];
}

// ---------------------------------------------------------------------------
// Stage one (b,n) X tile (64 rows x 256 f32) into LDS, fully coalesced,
// via async global->LDS. Rows padded to XROW_DW DWORDs.
// ---------------------------------------------------------------------------
__device__ __forceinline__ void stage_xtile(const float* __restrict__ X,
                                            uint32_t xbase, int b, int n,
                                            int wave, int lane) {
#pragma unroll
    for (int it = 0; it < 8; ++it) {
        const int t = wave * 8 + it;
        const float* gsrc = X + ((size_t)(b * T_ + t) * N_ + n) * F_ + lane * 4;
        const uint32_t ldst = xbase + (uint32_t)(t * (XROW_DW * 4) + lane * 16);
        async_copy_b128(ldst, gsrc);                 // bytes [lane*16, +16)
        async_copy_b128(ldst + 512, gsrc + 128);     // bytes [512+lane*16, +16)
    }
}

// ---------------------------------------------------------------------------
// Per-wave GEMM over K=256: A from LDS tile (this lane's row), B from packed
// f16 fragments. Produces this wave's 8 N-tile accumulators.
// ---------------------------------------------------------------------------
__device__ __forceinline__ void gemm_256(const float* __restrict__ xrow,
                                         const _Float16* __restrict__ WP,
                                         int lane, int nhalf, v8f acc[8]) {
    const int kb = (lane >> 4) * 8;  // A layout: K chunks {kb..kb+7, kb+16..kb+23}
#pragma unroll
    for (int s = 0; s < KSTEPS; ++s) {
        const float* xp = xrow + s * 32 + kb;
        v16h a;
#pragma unroll
        for (int i = 0; i < 8; ++i) {
            a[i]     = (_Float16)xp[i];
            a[i + 8] = (_Float16)xp[16 + i];
        }
#pragma unroll
        for (int j = 0; j < 8; ++j) {
            v16h bfr = *reinterpret_cast<const v16h*>(WP + wpk_off(s, nhalf * 8 + j, lane));
            acc[j] = __builtin_amdgcn_wmma_f32_16x16x32_f16(
                false, a, false, bfr, (short)0, acc[j], false, false);
        }
    }
}

// ---------------------------------------------------------------------------
// Pass 1: per (b,n): tr_h = gelu(X Wh + bh) for all 64 t; mean over t -> Hmean.
// ---------------------------------------------------------------------------
__global__ __launch_bounds__(256)
void pass1_kernel(const float* __restrict__ X, const _Float16* __restrict__ WhP,
                  const float* __restrict__ bh, float* __restrict__ Hmean) {
    __shared__ __attribute__((aligned(16))) float xtile[T_ * XROW_DW];
    __shared__ float hsum[F_];
    const int n = blockIdx.x, b = blockIdx.y;
    const int tid = threadIdx.x;
    const int lane = tid & 31, wave = tid >> 5;
    const int mt = wave & 3, nhalf = wave >> 2;

    stage_xtile(X, (uint32_t)(uintptr_t)xtile, b, n, wave, lane);
    hsum[tid] = 0.0f;
    wait_asynccnt0();
    __syncthreads();

    const float* xrow = xtile + (mt * 16 + (lane & 15)) * XROW_DW;

    v8f acc[8];
#pragma unroll
    for (int j = 0; j < 8; ++j)
#pragma unroll
        for (int r = 0; r < 8; ++r) acc[j][r] = 0.0f;

    gemm_256(xrow, WhP, lane, nhalf, acc);

#pragma unroll
    for (int j = 0; j < 8; ++j) {
        const int col = nhalf * 128 + j * 16 + (lane & 15);
        const float bias = bh[col];
        float csum = 0.0f;
#pragma unroll
        for (int r = 0; r < 8; ++r)
            csum += gelu_exact(acc[j][r] + bias);
        csum += __shfl_xor(csum, 16);          // rows 0-7 + rows 8-15
        if (lane < 16) atomicAdd(&hsum[col], csum);
    }
    __syncthreads();
    Hmean[((size_t)(b * N_ + n)) * F_ + tid] = hsum[tid] * (1.0f / T_);
}

// ---------------------------------------------------------------------------
// tr_a = relu(Hmean Wa + ba): 3312 rows = 207 M-tiles of 16.
// Wave w owns N-tiles {w, w+8}.
// ---------------------------------------------------------------------------
__global__ __launch_bounds__(256)
void tra_kernel(const float* __restrict__ Hmean, const _Float16* __restrict__ WaP,
                const float* __restrict__ ba, float* __restrict__ Ta) {
    const int tile = blockIdx.x;               // 0..206
    const int tid = threadIdx.x;
    const int lane = tid & 31, wave = tid >> 5;
    const int row = tile * 16 + (lane & 15);
    const float* arow = Hmean + (size_t)row * F_;
    const int kb = (lane >> 4) * 8;

    v8f acc0, acc1;
#pragma unroll
    for (int r = 0; r < 8; ++r) { acc0[r] = 0.0f; acc1[r] = 0.0f; }

#pragma unroll
    for (int s = 0; s < KSTEPS; ++s) {
        const float* xp = arow + s * 32 + kb;
        v16h a;
#pragma unroll
        for (int i = 0; i < 8; ++i) {
            a[i]     = (_Float16)xp[i];
            a[i + 8] = (_Float16)xp[16 + i];
        }
        v16h b0 = *reinterpret_cast<const v16h*>(WaP + wpk_off(s, wave, lane));
        v16h b1 = *reinterpret_cast<const v16h*>(WaP + wpk_off(s, wave + 8, lane));
        acc0 = __builtin_amdgcn_wmma_f32_16x16x32_f16(false, a, false, b0, (short)0, acc0, false, false);
        acc1 = __builtin_amdgcn_wmma_f32_16x16x32_f16(false, a, false, b1, (short)0, acc1, false, false);
    }

#pragma unroll
    for (int h = 0; h < 2; ++h) {
        const int j = wave + h * 8;
        const int col = j * 16 + (lane & 15);
        const float bias = ba[col];
#pragma unroll
        for (int r = 0; r < 8; ++r) {
            const int grow = tile * 16 + (lane >> 4) * 8 + r;
            float v = (h == 0 ? acc0[r] : acc1[r]) + bias;
            Ta[(size_t)grow * F_ + col] = fmaxf(v, 0.0f);
        }
    }
}

// ---------------------------------------------------------------------------
// Pass 2: recompute tr_h, attention = sigmoid(tr_h . tr_a), gated mean over t.
// ---------------------------------------------------------------------------
__global__ __launch_bounds__(256)
void pass2_kernel(const float* __restrict__ X, const _Float16* __restrict__ WhP,
                  const float* __restrict__ bh, const float* __restrict__ Ta,
                  float* __restrict__ out) {
    __shared__ __attribute__((aligned(16))) float xtile[T_ * XROW_DW];
    __shared__ float attL[T_];
    __shared__ float resL[F_];
    const int n = blockIdx.x, b = blockIdx.y;
    const int tid = threadIdx.x;
    const int lane = tid & 31, wave = tid >> 5;
    const int mt = wave & 3, nhalf = wave >> 2;

    stage_xtile(X, (uint32_t)(uintptr_t)xtile, b, n, wave, lane);
    resL[tid] = 0.0f;
    if (tid < T_) attL[tid] = 0.0f;
    wait_asynccnt0();
    __syncthreads();

    const float* xrow = xtile + (mt * 16 + (lane & 15)) * XROW_DW;

    v8f acc[8];
#pragma unroll
    for (int j = 0; j < 8; ++j)
#pragma unroll
        for (int r = 0; r < 8; ++r) acc[j][r] = 0.0f;

    gemm_256(xrow, WhP, lane, nhalf, acc);

    // bias + gelu in-place; fetch tr_a for this lane's column of each tile
    float ta[8];
#pragma unroll
    for (int j = 0; j < 8; ++j) {
        const int col = nhalf * 128 + j * 16 + (lane & 15);
        const float bias = bh[col];
        ta[j] = Ta[((size_t)(b * N_ + n)) * F_ + col];
#pragma unroll
        for (int r = 0; r < 8; ++r)
            acc[j][r] = gelu_exact(acc[j][r] + bias);
    }

    // per-row dot(tr_h, tr_a): lane sums its 8 cols, xor-reduce over 16 lanes
    float dotv[8];
#pragma unroll
    for (int r = 0; r < 8; ++r) dotv[r] = 0.0f;
#pragma unroll
    for (int j = 0; j < 8; ++j)
#pragma unroll
        for (int r = 0; r < 8; ++r) dotv[r] += acc[j][r] * ta[j];
#pragma unroll
    for (int r = 0; r < 8; ++r) {
        dotv[r] += __shfl_xor(dotv[r], 1);
        dotv[r] += __shfl_xor(dotv[r], 2);
        dotv[r] += __shfl_xor(dotv[r], 4);
        dotv[r] += __shfl_xor(dotv[r], 8);
    }
    const int rbase = mt * 16 + (lane >> 4) * 8;
    if ((lane & 15) == 0) {
#pragma unroll
        for (int r = 0; r < 8; ++r) atomicAdd(&attL[rbase + r], dotv[r]);
    }
    __syncthreads();
    if (tid < T_) attL[tid] = 1.0f / (1.0f + expf(-attL[tid]));
    __syncthreads();

    // gated accumulation: sum_t att[t] * tr_h[t][col]
    float att[8];
#pragma unroll
    for (int r = 0; r < 8; ++r) att[r] = attL[rbase + r];
#pragma unroll
    for (int j = 0; j < 8; ++j) {
        float cs = 0.0f;
#pragma unroll
        for (int r = 0; r < 8; ++r) cs += att[r] * acc[j][r];
        cs += __shfl_xor(cs, 16);
        if (lane < 16) atomicAdd(&resL[nhalf * 128 + j * 16 + lane], cs);
    }
    __syncthreads();
    out[((size_t)(b * N_ + n)) * F_ + tid] = resL[tid] * (1.0f / T_);
}

// ---------------------------------------------------------------------------
extern "C" void kernel_launch(void* const* d_in, const int* in_sizes, int n_in,
                              void* d_out, int out_size, void* d_ws, size_t ws_size,
                              hipStream_t stream) {
    const float* X  = (const float*)d_in[0];
    const float* Wh = (const float*)d_in[1];
    const float* bh = (const float*)d_in[2];
    const float* Wa = (const float*)d_in[3];
    const float* ba = (const float*)d_in[4];
    float* out = (float*)d_out;

    char* ws = (char*)d_ws;
    _Float16* WhP = (_Float16*)ws;                       // 65536 halves = 128 KB
    _Float16* WaP = (_Float16*)(ws + (128u << 10));      // 128 KB
    float* Hmean  = (float*)(ws + (256u << 10));         // 3312*256*4 = 3,391,488 B
    float* Ta     = (float*)(ws + (256u << 10) + 3391488u);
    // total workspace: ~6.9 MB

    hipLaunchKernelGGL(pack_w_kernel, dim3(64), dim3(256), 0, stream, Wh, Wa, WhP, WaP);
    hipLaunchKernelGGL(pass1_kernel, dim3(N_, B_), dim3(256), 0, stream, X, WhP, bh, Hmean);
    hipLaunchKernelGGL(tra_kernel, dim3(N_), dim3(256), 0, stream, Hmean, WaP, ba, Ta);
    hipLaunchKernelGGL(pass2_kernel, dim3(N_, B_), dim3(256), 0, stream, X, WhP, bh, Ta, out);
}
